// Phi3FastAttention_71262097375265
// MI455X (gfx1250) — matrix operand order repaired and checked by reading, not compiled, not run on hardware
//
#include <hip/hip_runtime.h>
#include <hip/hip_bf16.h>
#include <cstdint>
#include <cstddef>
#include <type_traits>

// ---------------------------------------------------------------------------
// Phi3 attention layer for gfx1250 (MI455X): bf16 WMMA everywhere.
//   B=2, S=2048, H=3072, nh=32, nkv=8, hd=96, causal, RoPE base 10000.
// Pipeline:
//   0) one-shot f32->bf16 conversion of hidden / w_qkv / w_o
//   1) QKV projection GEMM (bf16 in, bf16 out)
//   2) RoPE + repack (Q scaled, V transposed)
//   3) causal flash attention, TRANSPOSED form (S^T = K*Q^T, O^T = V^T*P^T):
//      queries live on lanes -> scalar softmax state, no LDS, shuffle exchange
//   4) O-projection GEMM (bf16 in, f32 out)
// ---------------------------------------------------------------------------

typedef __attribute__((ext_vector_type(16))) __bf16          bf16x16;
typedef __attribute__((ext_vector_type(8)))  float           floatx8;
typedef __attribute__((ext_vector_type(4)))  unsigned short  ushort4v;
typedef __attribute__((ext_vector_type(8)))  unsigned short  ushort8v;

constexpr int kB    = 2;
constexpr int kS    = 2048;
constexpr int kH    = 3072;
constexpr int kNH   = 32;
constexpr int kNKV  = 8;
constexpr int kHD   = 96;
constexpr int kQKV  = kNH * kHD + 2 * kNKV * kHD;   // 4608
constexpr int kM    = kB * kS;                      // 4096 tokens

__device__ __forceinline__ unsigned short f32_to_bf16(float f) {
  unsigned int x = __float_as_uint(f);
  x += 0x7FFFu + ((x >> 16) & 1u);          // round-to-nearest-even
  return (unsigned short)(x >> 16);
}
__device__ __forceinline__ float bf16_to_f32(unsigned short u) {
  return __uint_as_float(((unsigned int)u) << 16);
}

union FragU { bf16x16 v; unsigned short u[16]; };

// ---- A-operand (16x32, 16-bit) per-lane layout:
//   lanes 0-15 : row = lane,    K = c0+0..15
//   lanes 16-31: row = lane-16, K = c0+16..31
__device__ __forceinline__ bf16x16 load_a_bf16(const unsigned short* __restrict__ base, int ld,
                                               int row0, int c0, int lane) {
  const int hf = lane >> 4, r = lane & 15;
  const unsigned short* p = base + (size_t)(row0 + r) * ld + c0 + hf * 16;
  FragU f;
#pragma unroll
  for (int i = 0; i < 8; ++i) f.u[i]     = p[i];
#pragma unroll
  for (int i = 0; i < 8; ++i) f.u[8 + i] = p[8 + i];
  return f.v;
}

// ---- B-operand (32x16, 16-bit): lane holds column (col0 + lane&15);
//   lanes 0-15 get K = k0+0..15, lanes 16-31 get K = k0+16..31 (contiguous 32B).
//   Source array is row-major [Ncols x K] (i.e. the transposed operand).
__device__ __forceinline__ bf16x16 load_b_bf16(const unsigned short* __restrict__ base, int ld,
                                               int col0, int k0, int lane) {
  const int hf = lane >> 4, n = lane & 15;
  const unsigned short* p = base + (size_t)(col0 + n) * ld + k0 + hf * 16;
  FragU f;
#pragma unroll
  for (int i = 0; i < 16; ++i) f.u[i] = p[i];
  return f.v;
}

__device__ __forceinline__ floatx8 wmma_bf16(bf16x16 a, bf16x16 b, floatx8 c) {
  return __builtin_amdgcn_wmma_f32_16x16x32_bf16(false, a, false, b, (short)0, c, false, false);
}

// ---------------------------------------------------------------------------
// One-shot f32 -> bf16 conversion (4 elements / thread, vectorized).
// ---------------------------------------------------------------------------
__global__ __launch_bounds__(256) void cvt_f32_bf16(const float* __restrict__ in,
                                                    unsigned short* __restrict__ out,
                                                    size_t n) {
  const size_t i = ((size_t)blockIdx.x * 256 + threadIdx.x) * 4;
  if (i + 3 < n) {
    const float4 f = *(const float4*)(in + i);
    ushort4v o;
    o.x = f32_to_bf16(f.x); o.y = f32_to_bf16(f.y);
    o.z = f32_to_bf16(f.z); o.w = f32_to_bf16(f.w);
    *(ushort4v*)(out + i) = o;
  } else {
    for (size_t j = i; j < n; ++j) out[j] = f32_to_bf16(in[j]);
  }
}

// ---------------------------------------------------------------------------
// NT GEMM:  C[M,N] = A[M,K] * W[N,K]^T, both operands pre-converted bf16.
// 128 threads = 4 waves; each wave computes a 32x64 tile (8 WMMA / 6 frag loads);
// block covers 64x128.
// ---------------------------------------------------------------------------
template <bool OUT_BF16>
__global__ __launch_bounds__(128) void wmma_gemm_nt(const unsigned short* __restrict__ A,
                                                    const unsigned short* __restrict__ W,
                                                    void* __restrict__ Cp,
                                                    int M, int N, int K) {
  const int lane = threadIdx.x & 31;
  const int wave = threadIdx.x >> 5;
  const int m0 = blockIdx.y * 64 + (wave >> 1) * 32;
  const int n0 = blockIdx.x * 128 + (wave & 1) * 64;

  floatx8 c[2][4] = {};
  for (int k0 = 0; k0 < K; k0 += 32) {
    const bf16x16 a0 = load_a_bf16(A, K, m0, k0, lane);
    const bf16x16 a1 = load_a_bf16(A, K, m0 + 16, k0, lane);
#pragma unroll
    for (int j = 0; j < 4; ++j) {
      const bf16x16 b = load_b_bf16(W, K, n0 + j * 16, k0, lane);
      c[0][j] = wmma_bf16(a0, b, c[0][j]);
      c[1][j] = wmma_bf16(a1, b, c[1][j]);
    }
  }

  // C/D layout: lane holds column n = lane&15; rows v + 8*(lane>>4).
  const int n = lane & 15, mr = (lane >> 4) * 8;
#pragma unroll
  for (int i = 0; i < 2; ++i)
#pragma unroll
    for (int v = 0; v < 8; ++v) {
      const size_t r = (size_t)(m0 + i * 16 + mr + v) * N;
#pragma unroll
      for (int j = 0; j < 4; ++j) {
        if constexpr (OUT_BF16)
          ((unsigned short*)Cp)[r + n0 + j * 16 + n] = f32_to_bf16(c[i][j][v]);
        else
          ((float*)Cp)[r + n0 + j * 16 + n] = c[i][j][v];
      }
    }
}

// ---------------------------------------------------------------------------
// RoPE + repack: qkv[bs, 4608] (bf16) ->
//   Qb[b, nh, s, hd] (rope, * 1/sqrt(hd)),  Kb[b, nkv, s, hd] (rope),
//   Vt[b, nkv, hd, s] (transposed: PV A-fragments load it directly).
// ---------------------------------------------------------------------------
__global__ __launch_bounds__(256) void rope_pack(const unsigned short* __restrict__ qkv,
                                                 const int* __restrict__ pos,
                                                 unsigned short* __restrict__ Qb,
                                                 unsigned short* __restrict__ Kb,
                                                 unsigned short* __restrict__ Vt) {
  const size_t idx = (size_t)blockIdx.x * 256 + threadIdx.x;
  const size_t total = (size_t)kM * kQKV;
  if (idx >= total) return;
  const int col = (int)(idx % kQKV);
  const size_t m = idx / kQKV;
  const int b = (int)(m / kS), s = (int)(m % kS);
  const float x = bf16_to_f32(qkv[idx]);

  if (col >= kNH * kHD + kNKV * kHD) {            // ---- V: no rope, transpose
    const int cv = col - (kNH * kHD + kNKV * kHD);
    const int hv = cv / kHD, d = cv % kHD;
    Vt[((size_t)(b * kNKV + hv) * kHD + d) * kS + s] = f32_to_bf16(x);
    return;
  }
  const int d = col % kHD;                        // 3072 % 96 == 0 -> valid for Q and K
  const int j = (d < 48) ? d : d - 48;
  const float inv_freq = __powf(10000.0f, -(float)(2 * j) * (1.0f / (float)kHD));
  const float fr = (float)pos[m] * inv_freq;
  float sn, cs;
  __sincosf(fr, &sn, &cs);
  const float part = bf16_to_f32(qkv[idx - d + ((d < 48) ? d + 48 : d - 48)]);
  float y = x * cs + ((d < 48) ? -part : part) * sn;

  if (col < kNH * kHD) {                          // ---- Q (scale folded in)
    const int h = col / kHD;
    y *= 0.10206207262f;                          // 1/sqrt(96)
    Qb[((size_t)(b * kNH + h) * kS + s) * kHD + d] = f32_to_bf16(y);
  } else {                                        // ---- K
    const int hk = col / kHD - kNH;
    Kb[((size_t)(b * kNKV + hk) * kS + s) * kHD + d] = f32_to_bf16(y);
  }
}

// ---------------------------------------------------------------------------
// Causal flash attention, transposed form. 4 waves/block, 16 queries/wave,
// 32 keys/iteration. Lane n (and its xor-16 partner) own query q0+n:
//   S^T = K * Q^T        (A = K rows, B = Q rows)  -> scores per-query in-lane
//   O^T = V^T * P^T      (A = Vt rows = channels, B = P^T via 8 shfl_xor)
// Scalar per-lane softmax state; no LDS; exactly one masked diagonal chunk.
// ---------------------------------------------------------------------------
__global__ __launch_bounds__(128) void flash_attn(const unsigned short* __restrict__ Qb,
                                                  const unsigned short* __restrict__ Kb,
                                                  const unsigned short* __restrict__ Vt,
                                                  unsigned short* __restrict__ attn) {
  const int lane = threadIdx.x & 31;
  const int wave = threadIdx.x >> 5;
  const int bh = blockIdx.y;
  const int b = bh / kNH, h = bh % kNH, hk = h / (kNH / kNKV);
  const int q0 = blockIdx.x * 64 + wave * 16;
  const unsigned short* Qh = Qb + (size_t)(b * kNH + h) * kS * kHD;
  const unsigned short* Kh = Kb + (size_t)(b * kNKV + hk) * kS * kHD;
  const unsigned short* Vh = Vt + (size_t)(b * kNKV + hk) * kHD * kS;
  const int n = lane & 15, hf = lane >> 4;

  // Q as B-operand (loop invariant): column = query q0+n, contiguous hd chunks
  const bf16x16 qf0 = load_b_bf16(Qh, kHD, q0, 0, lane);
  const bf16x16 qf1 = load_b_bf16(Qh, kHD, q0, 32, lane);
  const bf16x16 qf2 = load_b_bf16(Qh, kHD, q0, 64, lane);

  floatx8 ot[6] = {};                 // O^T tiles: channel = t*16 + v + 8*hf, query = q0+n
  float mq = -__builtin_inff();       // per-query running max (scalar per lane)
  float lq = 0.f;                     // per-query running sum

  auto process_chunk = [&](int k0, auto masked_tag) {
    constexpr bool MASKED = decltype(masked_tag)::value;
    // ---- S^T: two 16x16 tiles; tile0 = keys [k0,k0+16), tile1 = [k0+16,k0+32)
    floatx8 s0 = {}, s1 = {};
    s0 = wmma_bf16(load_a_bf16(Kh, kHD, k0, 0, lane),       qf0, s0);
    s0 = wmma_bf16(load_a_bf16(Kh, kHD, k0, 32, lane),      qf1, s0);
    s0 = wmma_bf16(load_a_bf16(Kh, kHD, k0, 64, lane),      qf2, s0);
    s1 = wmma_bf16(load_a_bf16(Kh, kHD, k0 + 16, 0, lane),  qf0, s1);
    s1 = wmma_bf16(load_a_bf16(Kh, kHD, k0 + 16, 32, lane), qf1, s1);
    s1 = wmma_bf16(load_a_bf16(Kh, kHD, k0 + 16, 64, lane), qf2, s1);

    // ---- (diagonal chunk only) causal mask: key index = k0 + v + 8*hf (+16 for s1)
    if constexpr (MASKED) {
      const int qg = q0 + n;
#pragma unroll
      for (int v = 0; v < 8; ++v) {
        if (k0 + v + 8 * hf > qg)      s0[v] = -__builtin_inff();
        if (k0 + 16 + v + 8 * hf > qg) s1[v] = -__builtin_inff();
      }
    }

    // ---- per-query max: 16 in-lane values + xor-16 partner half
    float mx = fmaxf(s0[0], s1[0]);
#pragma unroll
    for (int v = 1; v < 8; ++v) mx = fmaxf(mx, fmaxf(s0[v], s1[v]));
    mx = fmaxf(mx, __shfl_xor(mx, 16, 32));

    const float mn = fmaxf(mq, mx);
    const float corr = __expf(mq - mn);
    mq = mn;

    // ---- exponentials + per-query sum (scalar state)
    float p0[8], p1[8], sum = 0.f;
#pragma unroll
    for (int v = 0; v < 8; ++v) {
      p0[v] = __expf(s0[v] - mn);
      p1[v] = __expf(s1[v] - mn);
      sum += p0[v] + p1[v];
    }
    sum += __shfl_xor(sum, 16, 32);
    lq = lq * corr + sum;
#pragma unroll
    for (int t = 0; t < 6; ++t)
#pragma unroll
      for (int v = 0; v < 8; ++v) ot[t][v] *= corr;

    // ---- assemble P^T B-operand: lane needs the partner half's 8 keys
    //   low lane n : u[0..7] = own p0 (keys k0+0..7),  u[8..15] = partner p0 (keys k0+8..15)
    //   high lane  : u[0..7] = partner p1 (k0+16..23), u[8..15] = own p1 (k0+24..31)
    FragU pf;
#pragma unroll
    for (int v = 0; v < 8; ++v) {
      const float send = hf ? p0[v] : p1[v];
      const float rec  = __shfl_xor(send, 16, 32);
      pf.u[v]     = f32_to_bf16(hf ? rec : p0[v]);
      pf.u[8 + v] = f32_to_bf16(hf ? p1[v] : rec);
    }

    // ---- O^T += V^T * P^T  (A = Vt rows = hd channels, contiguous keys)
#pragma unroll
    for (int t = 0; t < 6; ++t)
      ot[t] = wmma_bf16(load_a_bf16(Vh, kS, t * 16, k0, lane), pf.v, ot[t]);
  };

  // full (unmasked) chunks: keys k0..k0+31 all <= q0  <=>  k0+31 <= q0
  int k0 = 0;
  for (; k0 + 31 <= q0; k0 += 32) process_chunk(k0, std::false_type{});
  // exactly one diagonal (masked) chunk remains, covering up to kend = q0+15
  process_chunk(k0, std::true_type{});

  // ---- normalize, emit attn[token q0+n, h*96 + channel]: 8 contiguous bf16/tile
  const float inv = 1.0f / lq;
  unsigned short* dst =
      attn + ((size_t)b * kS + q0 + n) * (size_t)(kNH * kHD) + h * kHD + 8 * hf;
#pragma unroll
  for (int t = 0; t < 6; ++t) {
    ushort8v o;
#pragma unroll
    for (int v = 0; v < 8; ++v) o[v] = f32_to_bf16(ot[t][v] * inv);
    *(ushort8v*)(dst + t * 16) = o;
  }
}

// ---------------------------------------------------------------------------
extern "C" void kernel_launch(void* const* d_in, const int* in_sizes, int n_in,
                              void* d_out, int out_size, void* d_ws, size_t ws_size,
                              hipStream_t stream) {
  const float* hidden = (const float*)d_in[0];
  const int*   pos    = (const int*)d_in[1];
  const float* w_qkv  = (const float*)d_in[2];
  const float* w_o    = (const float*)d_in[3];
  float* out = (float*)d_out;

  char* ws = (char*)d_ws;
  size_t off = 0;
  auto alloc = [&](size_t bytes) -> void* {
    void* p = ws + off;
    off += (bytes + 255) & ~(size_t)255;
    return p;
  };
  const size_t nHid = (size_t)kM * kH;       // 12.58M
  const size_t nWq  = (size_t)kQKV * kH;     // 14.16M
  const size_t nWo  = (size_t)kH * kH;       //  9.44M
  unsigned short* hid_bf  = (unsigned short*)alloc(nHid * 2);                       // 25.2 MB
  unsigned short* wqkv_bf = (unsigned short*)alloc(nWq * 2);                        // 28.3 MB
  unsigned short* wo_bf   = (unsigned short*)alloc(nWo * 2);                        // 18.9 MB
  unsigned short* qkv     = (unsigned short*)alloc((size_t)kM * kQKV * 2);          // 37.7 MB
  unsigned short* Qb      = (unsigned short*)alloc((size_t)kB * kNH  * kS * kHD*2); // 25.2 MB
  unsigned short* Kb      = (unsigned short*)alloc((size_t)kB * kNKV * kS * kHD*2); //  6.3 MB
  unsigned short* Vt      = (unsigned short*)alloc((size_t)kB * kNKV * kHD * kS*2); //  6.3 MB
  unsigned short* attn    = (unsigned short*)alloc((size_t)kM * kH * 2);            // 25.2 MB

  // 0) one-shot precision conversion (keeps GEMM inner loops load+wmma only)
  cvt_f32_bf16<<<dim3((unsigned)((nHid / 4 + 255) / 256)), 256, 0, stream>>>(hidden, hid_bf, nHid);
  cvt_f32_bf16<<<dim3((unsigned)((nWq  / 4 + 255) / 256)), 256, 0, stream>>>(w_qkv, wqkv_bf, nWq);
  cvt_f32_bf16<<<dim3((unsigned)((nWo  / 4 + 255) / 256)), 256, 0, stream>>>(w_o, wo_bf, nWo);

  // 1) QKV projection: [4096,3072] x [4608,3072]^T -> bf16 qkv
  wmma_gemm_nt<true><<<dim3(kQKV / 128, kM / 64), 128, 0, stream>>>(
      hid_bf, wqkv_bf, qkv, kM, kQKV, kH);

  // 2) RoPE + repack (Q scaled by 1/sqrt(hd); V transposed)
  {
    const size_t total = (size_t)kM * kQKV;
    rope_pack<<<dim3((unsigned)((total + 255) / 256)), 256, 0, stream>>>(
        qkv, pos, Qb, Kb, Vt);
  }

  // 3) causal flash attention (GQA 4:1), transposed/LDS-free
  flash_attn<<<dim3(kS / 64, kB * kNH), 128, 0, stream>>>(Qb, Kb, Vt, attn);

  // 4) output projection: [4096,3072] x [3072,3072]^T -> f32 out
  wmma_gemm_nt<false><<<dim3(kH / 128, kM / 64), 128, 0, stream>>>(
      attn, wo_bf, out, kM, kH, kH);
}
